// GraphAutoencoder_33706903339598
// MI455X (gfx1250) — compile-verified, matrix operand order checked by reading, not run on hardware
//
#include <hip/hip_runtime.h>
#include <hip/hip_bf16.h>
#include <math.h>

// ---------------- problem constants ----------------
#define NN   100000   // nodes
#define NE   400000   // edges
#define NG   2000     // graphs
#define EMB  300
#define EMBP 320      // padded (mult of 64)
#define HID  600
#define HIDP 640      // padded (mult of 64)
#define FEAT 256
#define NL   5

typedef __attribute__((ext_vector_type(16))) __bf16 v16bf;
typedef __attribute__((ext_vector_type(8)))  float  v8f;

union FragBF { unsigned u[8]; v16bf v; };

__device__ __forceinline__ unsigned short f2bf(float x) {
  unsigned u = __float_as_uint(x);
  unsigned r = u + 0x7FFFu + ((u >> 16) & 1u);   // RNE
  return (unsigned short)(r >> 16);
}

__device__ __forceinline__ float act_f(float v, int act) {
  if (act == 1) return v > 0.f ? v : 0.f;                       // relu
  if (act == 2) return (v > 20.f) ? v : log1pf(expf(v));        // softplus
  return v;
}

// ---------------- elementwise / prep kernels ----------------

__global__ void zero_f32(float* p, int n) {
  int i = blockIdx.x * blockDim.x + threadIdx.x;
  if (i < n) p[i] = 0.f;
}

// h[i][f..f+3] = x_emb1[x[i,0]] + x_emb2[x[i,1]]   (pads -> 0)
__global__ void embed_v4_kernel(const int* __restrict__ x,
                                const float* __restrict__ xe1,
                                const float* __restrict__ xe2,
                                float* __restrict__ h) {
  int idx = blockIdx.x * blockDim.x + threadIdx.x;   // NN * EMBP/4
  if (idx >= NN * (EMBP / 4)) return;
  int i = idx / (EMBP / 4), f = (idx % (EMBP / 4)) * 4;
  float4 v = make_float4(0.f, 0.f, 0.f, 0.f);
  if (f < EMB) {
    int i0 = x[i * 2], i1 = x[i * 2 + 1];
    float4 a = *(const float4*)(xe1 + i0 * EMB + f);
    float4 b = *(const float4*)(xe2 + i1 * EMB + f);
    v = make_float4(a.x + b.x, a.y + b.y, a.z + b.z, a.w + b.w);
  }
  *(float4*)(h + (size_t)i * EMBP + f) = v;
}

// agg = h + self_loop_edge_emb (attr (4,0)); pads stay 0
__global__ void agg_init_v4_kernel(const float* __restrict__ h,
                                   const float* __restrict__ e1,
                                   const float* __restrict__ e2,
                                   float* __restrict__ agg, int l) {
  int idx = blockIdx.x * blockDim.x + threadIdx.x;
  if (idx >= NN * (EMBP / 4)) return;
  int i = idx / (EMBP / 4), f = (idx % (EMBP / 4)) * 4;
  float4 v = *(const float4*)(h + (size_t)i * EMBP + f);
  if (f < EMB) {
    float4 a = *(const float4*)(e1 + (l * 5 + 4) * EMB + f);
    float4 b = *(const float4*)(e2 + (l * 3 + 0) * EMB + f);
    v = make_float4(v.x + a.x + b.x, v.y + a.y + b.y,
                    v.z + a.z + b.z, v.w + a.w + b.w);
  }
  *(float4*)(agg + (size_t)i * EMBP + f) = v;
}

// per-edge message scatter, 4 features/thread: agg[dst] += h[src] + edge_emb
__global__ void scatter_v4_kernel(const float* __restrict__ h,
                                  const int* __restrict__ ei,
                                  const int* __restrict__ ea,
                                  const float* __restrict__ e1,
                                  const float* __restrict__ e2,
                                  float* __restrict__ agg, int l) {
  int idx = blockIdx.x * blockDim.x + threadIdx.x;   // NE * EMB/4
  if (idx >= NE * (EMB / 4)) return;
  int e = idx / (EMB / 4), f = (idx % (EMB / 4)) * 4;
  int src = ei[e], dst = ei[NE + e];
  int a0 = ea[e * 2], a1 = ea[e * 2 + 1];
  float4 hv = *(const float4*)(h + (size_t)src * EMBP + f);
  float4 ev1 = *(const float4*)(e1 + (l * 5 + a0) * EMB + f);
  float4 ev2 = *(const float4*)(e2 + (l * 3 + a1) * EMB + f);
  float* dp = agg + (size_t)dst * EMBP + f;
  atomicAdd(dp + 0, hv.x + ev1.x + ev2.x);
  atomicAdd(dp + 1, hv.y + ev1.y + ev2.y);
  atomicAdd(dp + 2, hv.z + ev1.z + ev2.z);
  atomicAdd(dp + 3, hv.w + ev1.w + ev2.w);
}

// weights f32 [K][N] -> bf16 transposed padded [Np][Kp] (zero pad)
__global__ void convert_wT(const float* __restrict__ w, int K, int N,
                           unsigned short* __restrict__ wt, int Kp, int Np) {
  int idx = blockIdx.x * blockDim.x + threadIdx.x;
  if (idx >= Np * Kp) return;
  int n = idx / Kp, k = idx % Kp;
  wt[idx] = (n < N && k < K) ? f2bf(w[k * N + n]) : (unsigned short)0;
}

__global__ void pad_bias(const float* __restrict__ b, int N,
                         float* __restrict__ bp, int Np) {
  int i = blockIdx.x * blockDim.x + threadIdx.x;
  if (i < Np) bp[i] = (i < N) ? b[i] : 0.f;
}

// ---------------- WMMA GEMM: C[M,Np] = act(A[M,Kp] * Bt^T + bias) ----------------
// AT: 0 = A is f32 (converted to bf16 while staging), 1 = A is bf16
// OT: 0 = C is f32, 1 = C is bf16
// Bt: bf16, N-major [Np][Kp] (pre-transposed, zero-padded)
// Requires: Np % 64 == 0, Kp % 32 == 0, A pad columns are zero in memory.
template <int AT, int OT>
__global__ __launch_bounds__(256) void wmma_gemm_kernel(
    const void* __restrict__ A_, int lda,
    const unsigned short* __restrict__ Bt,
    const float* __restrict__ biasPad,
    void* __restrict__ C_, int ldc,
    int M, int Kp, int Nstore, int act) {
  __shared__ unsigned Als[128][17];  // 128 rows x 16 K-pairs (+pad)
  __shared__ unsigned Bls[64][17];   // 64 cols  x 16 K-pairs (+pad)

  const int tid  = threadIdx.x;
  const int lane = tid & 31;
  const int wave = tid >> 5;
  const int waveM = wave & 3;   // 0..3 -> 32-row strips
  const int waveN = wave >> 2;  // 0..1 -> 32-col strips
  const int rowBase = blockIdx.x * 128;
  const int colBase = blockIdx.y * 64;
  const bool fullM = (rowBase + 128 <= M);  // uniform: skip row guards

  const v8f vzero = {0.f, 0.f, 0.f, 0.f, 0.f, 0.f, 0.f, 0.f};
  v8f acc[2][2];
  acc[0][0] = vzero; acc[0][1] = vzero; acc[1][0] = vzero; acc[1][1] = vzero;

  const float*          Af = (const float*)A_;
  const unsigned short* Ab = (const unsigned short*)A_;

  for (int kk = 0; kk < Kp; kk += 32) {
    __syncthreads();
    // ---- stage A tile (128 x 32 elems = 128 x 16 packed pairs) ----
    if (fullM) {
#pragma unroll
      for (int i = 0; i < 8; ++i) {
        int idx = i * 256 + tid;
        int r = idx >> 4, j = idx & 15;
        int gr = rowBase + r;
        unsigned pv;
        if (AT == 0) {
          const float2 p = *(const float2*)(Af + (size_t)gr * lda + kk + 2 * j);
          pv = (unsigned)f2bf(p.x) | ((unsigned)f2bf(p.y) << 16);
        } else {
          pv = *(const unsigned*)(Ab + (size_t)gr * lda + kk + 2 * j);
        }
        Als[r][j] = pv;
      }
    } else {
#pragma unroll
      for (int i = 0; i < 8; ++i) {
        int idx = i * 256 + tid;
        int r = idx >> 4, j = idx & 15;
        int gr = rowBase + r;
        unsigned pv = 0u;
        if (gr < M) {
          if (AT == 0) {
            const float2 p = *(const float2*)(Af + (size_t)gr * lda + kk + 2 * j);
            pv = (unsigned)f2bf(p.x) | ((unsigned)f2bf(p.y) << 16);
          } else {
            pv = *(const unsigned*)(Ab + (size_t)gr * lda + kk + 2 * j);
          }
        }
        Als[r][j] = pv;
      }
    }
    // ---- stage B tile (64 cols x 32 elems), coalesced along Kp ----
#pragma unroll
    for (int i = 0; i < 4; ++i) {
      int idx = i * 256 + tid;
      int n = idx >> 4, j = idx & 15;
      Bls[n][j] = *(const unsigned*)(Bt + (size_t)(colBase + n) * Kp + kk + 2 * j);
    }
    __syncthreads();

    // ---- gather fragments per ISA 16-bit A/B layout ----
    FragBF fa[2], fb[2];
    const int jb = (lane >> 4) << 2;  // lanes 16-31 take K=8..15 / 24..31 halves
#pragma unroll
    for (int s = 0; s < 2; ++s) {
      int m = waveM * 32 + s * 16 + (lane & 15);
      int n = waveN * 32 + s * 16 + (lane & 15);
#pragma unroll
      for (int t = 0; t < 8; ++t) {
        int jj = jb + t + ((t >> 2) << 2);  // t=0..3 -> jb+t ; t=4..7 -> jb+8..11
        fa[s].u[t] = Als[m][jj];
        fb[s].u[t] = Bls[n][jj];
      }
    }
#pragma unroll
    for (int sm = 0; sm < 2; ++sm)
#pragma unroll
      for (int sn = 0; sn < 2; ++sn)
        acc[sm][sn] = __builtin_amdgcn_wmma_f32_16x16x32_bf16(
            false, fa[sm].v, false, fb[sn].v, (short)0, acc[sm][sn], false, false);
  }

  // ---- epilogue: bias + activation + store (f32 or bf16) ----
#pragma unroll
  for (int sm = 0; sm < 2; ++sm) {
#pragma unroll
    for (int sn = 0; sn < 2; ++sn) {
      int n = colBase + waveN * 32 + sn * 16 + (lane & 15);
#pragma unroll
      for (int j = 0; j < 8; ++j) {
        int m = rowBase + waveM * 32 + sm * 16 + j + ((lane >> 4) << 3);
        if ((fullM || m < M) && n < Nstore) {
          float v = act_f(acc[sm][sn][j] + biasPad[n], act);
          if (OT == 0) ((float*)C_)[(size_t)m * ldc + n] = v;
          else         ((unsigned short*)C_)[(size_t)m * ldc + n] = f2bf(v);
        }
      }
    }
  }
}

// ---------------- BatchNorm (coalesced row-chunk partials) ----------------
// Each block streams 256 contiguous rows; thread t owns features t and t+256.
// Partial sums land in zeroed ssum/ssq via per-feature atomics.
__global__ __launch_bounds__(256) void bn_partial_kernel(
    const float* __restrict__ h, float* __restrict__ ssum, float* __restrict__ ssq) {
  int r0 = blockIdx.x * 256;
  int t = threadIdx.x;
  int f1 = t + 256;                    // valid only if < EMB (t < 44)
  int rend = NN - r0; if (rend > 256) rend = 256;
  float a0 = 0.f, q0 = 0.f, a1 = 0.f, q1 = 0.f;
  for (int r = 0; r < rend; ++r) {
    const float* row = h + (size_t)(r0 + r) * EMBP;
    float v = row[t];   a0 += v; q0 += v * v;
    if (f1 < EMB) { float w = row[f1]; a1 += w; q1 += w * w; }
  }
  atomicAdd(&ssum[t], a0); atomicAdd(&ssq[t], q0);
  if (f1 < EMB) { atomicAdd(&ssum[f1], a1); atomicAdd(&ssq[f1], q1); }
}

__global__ void bn_finalize_kernel(const float* __restrict__ ssum,
                                   const float* __restrict__ ssq,
                                   float* __restrict__ mean,
                                   float* __restrict__ rstd) {
  int f = blockIdx.x * blockDim.x + threadIdx.x;
  if (f >= EMB) return;
  float m = ssum[f] / (float)NN;
  float var = ssq[f] / (float)NN - m * m;
  mean[f] = m;
  rstd[f] = rsqrtf(var + 1e-5f);
}

__global__ void bn_apply_v4_kernel(float* __restrict__ h,
                                   const float* __restrict__ mean,
                                   const float* __restrict__ rstd,
                                   const float* __restrict__ gamma,
                                   const float* __restrict__ beta,
                                   int do_relu) {
  int idx = blockIdx.x * blockDim.x + threadIdx.x;
  if (idx >= NN * (EMBP / 4)) return;
  int i = idx / (EMBP / 4), f = (idx % (EMBP / 4)) * 4;
  if (f >= EMB) return;  // pads already 0
  float* p = h + (size_t)i * EMBP + f;
  float4 v = *(float4*)p;
  float4 mu = *(const float4*)(mean + f);
  float4 rs = *(const float4*)(rstd + f);
  float4 g  = *(const float4*)(gamma + f);
  float4 bb = *(const float4*)(beta + f);
  float4 o = make_float4((v.x - mu.x) * rs.x * g.x + bb.x,
                         (v.y - mu.y) * rs.y * g.y + bb.y,
                         (v.z - mu.z) * rs.z * g.z + bb.z,
                         (v.w - mu.w) * rs.w * g.w + bb.w);
  if (do_relu) {
    o.x = fmaxf(o.x, 0.f); o.y = fmaxf(o.y, 0.f);
    o.z = fmaxf(o.z, 0.f); o.w = fmaxf(o.w, 0.f);
  }
  *(float4*)p = o;
}

// ---------------- pooling ----------------
__global__ void count_kernel(const int* __restrict__ batch, float* __restrict__ counts) {
  int i = blockIdx.x * blockDim.x + threadIdx.x;
  if (i < NN) atomicAdd(&counts[batch[i]], 1.0f);
}

__global__ void pool_sum_v4_kernel(const float* __restrict__ h,
                                   const int* __restrict__ batch,
                                   float* __restrict__ sums) {
  int idx = blockIdx.x * blockDim.x + threadIdx.x;   // NN * EMB/4
  if (idx >= NN * (EMB / 4)) return;
  int i = idx / (EMB / 4), f = (idx % (EMB / 4)) * 4;
  float4 v = *(const float4*)(h + (size_t)i * EMBP + f);
  float* dp = sums + (size_t)batch[i] * EMBP + f;
  atomicAdd(dp + 0, v.x);
  atomicAdd(dp + 1, v.y);
  atomicAdd(dp + 2, v.z);
  atomicAdd(dp + 3, v.w);
}

__global__ void pool_div_kernel(const float* __restrict__ sums,
                                const float* __restrict__ counts,
                                float* __restrict__ pooled) {
  int idx = blockIdx.x * blockDim.x + threadIdx.x;
  if (idx >= NG * EMBP) return;
  int g = idx / EMBP;
  pooled[idx] = sums[idx] / fmaxf(counts[g], 1.0f);
}

// ---------------- host launcher ----------------
static inline int cdiv(int a, int b) { return (a + b - 1) / b; }

extern "C" void kernel_launch(void* const* d_in, const int* in_sizes, int n_in,
                              void* d_out, int out_size, void* d_ws, size_t ws_size,
                              hipStream_t stream) {
  // inputs (setup_inputs order)
  const int*   x     = (const int*)d_in[0];
  const int*   ei    = (const int*)d_in[1];
  const int*   ea    = (const int*)d_in[2];
  const int*   batch = (const int*)d_in[3];
  const float* xe1   = (const float*)d_in[4];
  const float* xe2   = (const float*)d_in[5];
  const float* ee1   = (const float*)d_in[6];
  const float* ee2   = (const float*)d_in[7];
  const float* w1    = (const float*)d_in[8];
  const float* b1    = (const float*)d_in[9];
  const float* w2    = (const float*)d_in[10];
  const float* b2    = (const float*)d_in[11];
  const float* gam   = (const float*)d_in[12];
  const float* bet   = (const float*)d_in[13];
  const float* dw1   = (const float*)d_in[14];
  const float* db1   = (const float*)d_in[15];
  const float* dw2   = (const float*)d_in[16];
  const float* db2   = (const float*)d_in[17];
  const float* hw1   = (const float*)d_in[18];
  const float* hb1   = (const float*)d_in[19];
  const float* hw2   = (const float*)d_in[20];
  const float* hb2   = (const float*)d_in[21];

  // workspace carve-up (256B aligned)
  char* ws = (char*)d_ws;
  size_t off = 0;
  auto take = [&](size_t bytes) -> char* {
    off = (off + 255) & ~(size_t)255;
    char* p = ws + off; off += bytes; return p;
  };
  float*          hbuf   = (float*)take((size_t)NN * EMBP * 4);
  float*          aggbuf = (float*)take((size_t)NN * EMBP * 4);
  unsigned short* hidbuf = (unsigned short*)take((size_t)NN * HIDP * 2);
  unsigned short* wt1    = (unsigned short*)take((size_t)HIDP * EMBP * 2);
  unsigned short* wt2    = (unsigned short*)take((size_t)EMBP * HIDP * 2);
  float*          b1p    = (float*)take(HIDP * 4);
  float*          b2p    = (float*)take(EMBP * 4);
  unsigned short* wtd1   = (unsigned short*)take((size_t)128 * EMBP * 2);
  float*          bd1p   = (float*)take(128 * 4);
  unsigned short* wtd2   = (unsigned short*)take((size_t)256 * 128 * 2);
  float*          bd2p   = (float*)take(256 * 4);
  unsigned short* wth1   = (unsigned short*)take((size_t)128 * 256 * 2);
  float*          bh1p   = (float*)take(128 * 4);
  unsigned short* wth2   = (unsigned short*)take((size_t)64 * 128 * 2);
  float*          bh2p   = (float*)take(64 * 4);
  float*          meanb  = (float*)take(EMBP * 4);
  float*          rstdb  = (float*)take(EMBP * 4);
  float*          ssum   = (float*)take(EMBP * 4);
  float*          ssq    = (float*)take(EMBP * 4);
  float*          sums   = (float*)take((size_t)NG * EMBP * 4);
  float*          counts = (float*)take(NG * 4);
  float*          pooled = (float*)take((size_t)NG * EMBP * 4);
  unsigned short* t1     = (unsigned short*)take((size_t)NG * 128 * 2);
  unsigned short* t2     = (unsigned short*)take((size_t)NG * 128 * 2);
  (void)ws_size; (void)in_sizes; (void)n_in; (void)out_size;

  const int B = 256;
  const int nV4 = NN * (EMBP / 4);   // 8M v4-elements

  // decoder weight prep (padded, transposed, bf16)
  convert_wT<<<cdiv(128 * EMBP, B), B, 0, stream>>>(dw1, EMB, 128, wtd1, EMBP, 128);
  pad_bias<<<1, 128, 0, stream>>>(db1, 128, bd1p, 128);
  convert_wT<<<cdiv(256 * 128, B), B, 0, stream>>>(dw2, 128, 256, wtd2, 128, 256);
  pad_bias<<<1, 256, 0, stream>>>(db2, 256, bd2p, 256);
  convert_wT<<<cdiv(128 * 256, B), B, 0, stream>>>(hw1, 256, 128, wth1, 256, 128);
  pad_bias<<<1, 128, 0, stream>>>(hb1, 128, bh1p, 128);
  convert_wT<<<cdiv(64 * 128, B), B, 0, stream>>>(hw2, 128, 2, wth2, 128, 64);
  pad_bias<<<1, 64, 0, stream>>>(hb2, 2, bh2p, 64);

  // node embedding
  embed_v4_kernel<<<cdiv(nV4, B), B, 0, stream>>>(x, xe1, xe2, hbuf);

  for (int l = 0; l < NL; ++l) {
    // layer weights -> bf16 transposed/padded
    convert_wT<<<cdiv(HIDP * EMBP, B), B, 0, stream>>>(w1 + (size_t)l * EMB * HID, EMB, HID, wt1, EMBP, HIDP);
    pad_bias<<<cdiv(HIDP, B), B, 0, stream>>>(b1 + l * HID, HID, b1p, HIDP);
    convert_wT<<<cdiv(EMBP * HIDP, B), B, 0, stream>>>(w2 + (size_t)l * HID * EMB, HID, EMB, wt2, HIDP, EMBP);
    pad_bias<<<cdiv(EMBP, B), B, 0, stream>>>(b2 + l * EMB, EMB, b2p, EMBP);

    // message passing
    agg_init_v4_kernel<<<cdiv(nV4, B), B, 0, stream>>>(hbuf, ee1, ee2, aggbuf, l);
    scatter_v4_kernel<<<cdiv(NE * (EMB / 4), B), B, 0, stream>>>(hbuf, ei, ea, ee1, ee2, aggbuf, l);

    // MLP: hid = relu(agg @ W1 + b1)  -> bf16
    {
      dim3 g(cdiv(NN, 128), HIDP / 64);
      wmma_gemm_kernel<0, 1><<<g, B, 0, stream>>>(aggbuf, EMBP, wt1, b1p,
                                                  hidbuf, HIDP, NN, EMBP, HIDP, /*relu*/1);
    }
    // h = hid @ W2 + b2  -> f32
    {
      dim3 g(cdiv(NN, 128), EMBP / 64);
      wmma_gemm_kernel<1, 0><<<g, B, 0, stream>>>(hidbuf, HIDP, wt2, b2p,
                                                  hbuf, EMBP, NN, HIDP, EMBP, /*none*/0);
    }
    // batchnorm (+relu except last layer)
    zero_f32<<<cdiv(EMBP, B), B, 0, stream>>>(ssum, EMBP);
    zero_f32<<<cdiv(EMBP, B), B, 0, stream>>>(ssq, EMBP);
    bn_partial_kernel<<<cdiv(NN, 256), 256, 0, stream>>>(hbuf, ssum, ssq);
    bn_finalize_kernel<<<cdiv(EMB, B), B, 0, stream>>>(ssum, ssq, meanb, rstdb);
    bn_apply_v4_kernel<<<cdiv(nV4, B), B, 0, stream>>>(hbuf, meanb, rstdb,
                                                       gam + l * EMB, bet + l * EMB,
                                                       (l < NL - 1) ? 1 : 0);
  }

  // global mean pool
  zero_f32<<<cdiv(NG * EMBP, B), B, 0, stream>>>(sums, NG * EMBP);
  zero_f32<<<cdiv(NG, B), B, 0, stream>>>(counts, NG);
  count_kernel<<<cdiv(NN, B), B, 0, stream>>>(batch, counts);
  pool_sum_v4_kernel<<<cdiv(NN * (EMB / 4), B), B, 0, stream>>>(hbuf, batch, sums);
  pool_div_kernel<<<cdiv(NG * EMBP, B), B, 0, stream>>>(sums, counts, pooled);

  float* dec = (float*)d_out;                     // [NG,256]
  float* out = (float*)d_out + (size_t)NG * FEAT; // [NG,2]

  // decoder: dec = relu(pooled @ dW1 + db1) @ dW2 + db2
  {
    dim3 g(cdiv(NG, 128), 128 / 64);
    wmma_gemm_kernel<0, 1><<<g, B, 0, stream>>>(pooled, EMBP, wtd1, bd1p,
                                                t1, 128, NG, EMBP, 128, /*relu*/1);
  }
  {
    dim3 g(cdiv(NG, 128), 256 / 64);
    wmma_gemm_kernel<1, 0><<<g, B, 0, stream>>>(t1, 128, wtd2, bd2p,
                                                dec, FEAT, NG, 128, FEAT, /*none*/0);
  }
  // head: out = softplus(dec @ hW1 + hb1) @ hW2 + hb2
  {
    dim3 g(cdiv(NG, 128), 128 / 64);
    wmma_gemm_kernel<0, 1><<<g, B, 0, stream>>>(dec, FEAT, wth1, bh1p,
                                                t2, 128, NG, FEAT, 128, /*softplus*/2);
  }
  {
    dim3 g(cdiv(NG, 128), 64 / 64);
    wmma_gemm_kernel<1, 0><<<g, B, 0, stream>>>(t2, 128, wth2, bh2p,
                                                out, 2, NG, 128, /*Nstore=*/2, /*none*/0);
  }
}
